// MultiViewEPSSClassifier_72834055406016
// MI455X (gfx1250) — compile-verified, matrix operand order checked by reading, not run on hardware
//
#include <hip/hip_runtime.h>
#include <hip/hip_bf16.h>
#include <math.h>

// ---------------- types ----------------
typedef __bf16 bf16_t;
typedef bf16_t v16bf __attribute__((ext_vector_type(16)));
typedef bf16_t v8bf  __attribute__((ext_vector_type(8)));
typedef float  v8f   __attribute__((ext_vector_type(8)));

#define HCH 128

// ---------------- conversion kernels ----------------
__global__ void cvt_bf16_kernel(const float* __restrict__ in, bf16_t* __restrict__ out, long n) {
  long i = (long)blockIdx.x * blockDim.x + threadIdx.x;
  if (i < n) out[i] = (bf16_t)in[i];
}

__global__ void fill_kernel(float* __restrict__ p, float v, long n) {
  long i = (long)blockIdx.x * blockDim.x + threadIdx.x;
  if (i < n) p[i] = v;
}

// ---------------- pack B (f32 -> bf16, WMMA fragment order) ----------------
// Logical B is K x Ncols row-major (if srcTransposed, source holds B^T: in[n*K + k]).
// Output fragment order: out[((kt*nTiles + nt)*32 + lane)*16 + e]
// where lane = half*16 + (n&15), and element e maps to K offset:
//   e<8 : kr = half*8 + e        e>=8 : kr = 16 + half*8 + (e-8)
__global__ void pack_b_kernel(const float* __restrict__ in, bf16_t* __restrict__ out,
                              int K, int Ncols, int srcTransposed) {
  long i = (long)blockIdx.x * blockDim.x + threadIdx.x;
  if (i >= (long)K * Ncols) return;
  int k = (int)(i / Ncols), n = (int)(i % Ncols);
  float v = srcTransposed ? in[(long)n * K + k] : in[i];
  int kt = k >> 5, kr = k & 31;
  int nt = n >> 4, nc = n & 15;
  int half = (kr >> 3) & 1;
  int e = (kr & 7) + ((kr & 16) ? 8 : 0);
  int lane = half * 16 + nc;
  int nTiles = Ncols >> 4;
  out[(((long)kt * nTiles + nt) * 32 + lane) * 16 + e] = (bf16_t)v;
}

// ---------------- WMMA bf16 GEMM ----------------
// C(MxN) = A(MxK) * Bpacked(KxN) [+ bias[N]]; A bf16 row-major, B pre-packed fragments.
// block = 256 threads = 8 waves; block covers 64 rows x 128 cols.
// wave w: M-tile (w>>1), 4 N-tiles starting at blockIdx.y*8 + (w&1)*4.
// K multiple of 32; Ncols multiple of 128 (grid.y = Ncols/128).
__global__ __launch_bounds__(256) void gemm_wmma_bf16_kernel(
    const bf16_t* __restrict__ A, const bf16_t* __restrict__ Bp,
    const float* __restrict__ bias, float* __restrict__ Cf, bf16_t* __restrict__ Cb,
    int M, int K, int Ncols)
{
  __shared__ bf16_t As[64][40];     // 64x32 tile, padded stride 40 (80B, keeps 16B align)
  const int t    = threadIdx.x;
  const int wave = t >> 5;
  const int lane = t & 31;
  const int half = lane >> 4;
  const int mr   = lane & 15;
  const int mt   = wave >> 1;       // M-tile within block (0..3)
  const int nh   = wave & 1;
  const int mBase  = blockIdx.x * 64;
  const int nTiles = Ncols >> 4;
  const int ntBase = blockIdx.y * 8 + nh * 4;   // first of this wave's 4 n-tiles
  const int kb     = half * 8;

  // staging coordinates: 16 contiguous bytes (8 bf16) per thread
  const int sar = t >> 2;                 // LDS row 0..63
  const int sac = (t & 3) * 8;            // LDS col offset {0,8,16,24}
  // clamp OOB rows to M-1: in-bounds read; contaminated rows are never stored
  const int sgrow = (mBase + sar < M) ? (mBase + sar) : (M - 1);
  const bf16_t* Arow = A + (long)sgrow * K + sac;

  v8f acc[4] = {v8f{}, v8f{}, v8f{}, v8f{}};

  for (int kk = 0; kk < K; kk += 32) {
    // stage A tile 64x32: one b128 global load + one b128 LDS store per thread
    *(v8bf*)&As[sar][sac] = *(const v8bf*)(Arow + kk);
    if (kk + 32 < K) __builtin_prefetch(Arow + kk + 32, 0, 3);
    __syncthreads();

    // A fragment: two contiguous 16B chunks per lane -> ds_load_b128 x2
    v16bf a;
    const int arow = mt * 16 + mr;
    #pragma unroll
    for (int j = 0; j < 8; ++j) {
      a[j]     = As[arow][kb + j];
      a[8 + j] = As[arow][16 + kb + j];
    }

    // B fragments: contiguous 32B per lane per tile (pre-packed) -> global_load_b128 x2
    const int  kt    = kk >> 5;
    const long fbase = ((long)kt * nTiles + ntBase) * 512 + (long)lane * 16;
    const v16bf* bp  = (const v16bf*)(Bp + fbase);   // tile stride = 512 elems = 32 v16bf

    acc[0] = __builtin_amdgcn_wmma_f32_16x16x32_bf16(false, a, false, bp[0],  (short)0, acc[0], false, false);
    acc[1] = __builtin_amdgcn_wmma_f32_16x16x32_bf16(false, a, false, bp[32], (short)0, acc[1], false, false);
    acc[2] = __builtin_amdgcn_wmma_f32_16x16x32_bf16(false, a, false, bp[64], (short)0, acc[2], false, false);
    acc[3] = __builtin_amdgcn_wmma_f32_16x16x32_bf16(false, a, false, bp[96], (short)0, acc[3], false, false);
    __syncthreads();
  }

  // C/D layout: VGPR r -> row = r (lanes 0-15) / 8+r (lanes 16-31), col = lane&15
  #pragma unroll
  for (int q = 0; q < 4; ++q) {
    int col = (ntBase + q) * 16 + mr;
    float bv = bias ? bias[col] : 0.f;
    #pragma unroll
    for (int r = 0; r < 8; ++r) {
      int row = mBase + mt * 16 + half * 8 + r;
      if (row < M) {
        float v = acc[q][r] + bv;
        if (Cf) Cf[(long)row * Ncols + col] = v;
        if (Cb) Cb[(long)row * Ncols + col] = (bf16_t)v;
      }
    }
  }
}

// ---------------- LayerNorm (+optional GELU, +optional residual) ----------------
__global__ __launch_bounds__(128) void ln_kernel(
    const float* __restrict__ in, const float* __restrict__ gw, const float* __restrict__ gb,
    const float* __restrict__ resid, float* __restrict__ outf, bf16_t* __restrict__ outb,
    long ostride, int act)
{
  const int n = blockIdx.x, t = threadIdx.x;
  __shared__ float red[128];
  float x = in[(long)n * HCH + t];
  red[t] = x; __syncthreads();
  #pragma unroll
  for (int s = 64; s > 0; s >>= 1) { if (t < s) red[t] += red[t + s]; __syncthreads(); }
  float mu = red[0] * (1.0f / HCH);
  __syncthreads();
  float d = x - mu;
  red[t] = d * d; __syncthreads();
  #pragma unroll
  for (int s = 64; s > 0; s >>= 1) { if (t < s) red[t] += red[t + s]; __syncthreads(); }
  float var = red[0] * (1.0f / HCH);
  float y = d * rsqrtf(var + 1e-5f) * gw[t] + gb[t];
  if (act == 1) y = 0.5f * y * (1.0f + erff(y * 0.7071067811865475f));  // exact GELU
  if (resid) y += resid[(long)n * HCH + t];
  outf[(long)n * ostride + t] = y;
  if (outb) outb[(long)n * ostride + t] = (bf16_t)y;
}

// ---------------- masked edge scatter-add ----------------
__global__ void scatter_kernel(const bf16_t* __restrict__ m, const int* __restrict__ src,
    const int* __restrict__ dst, const int* __restrict__ et, unsigned vmask,
    float* __restrict__ agg, int E)
{
  long tid = (long)blockIdx.x * blockDim.x + threadIdx.x;
  int e = (int)(tid >> 5);
  if (e >= E) return;
  int ty = et[e];
  if (!((vmask >> ty) & 1u)) return;
  int c = ((int)tid & 31) * 4;
  const bf16_t* mp = m + (long)src[e] * HCH + c;
  float* ap = agg + (long)dst[e] * HCH + c;
  atomicAdd(ap + 0, (float)mp[0]);
  atomicAdd(ap + 1, (float)mp[1]);
  atomicAdd(ap + 2, (float)mp[2]);
  atomicAdd(ap + 3, (float)mp[3]);
}

// ---------------- GRU cell gates (torch order r,z,n) ----------------
__global__ void gru_kernel(const float* __restrict__ gi, const float* __restrict__ gh,
                           float* __restrict__ hf, bf16_t* __restrict__ hb, long NH)
{
  long i = (long)blockIdx.x * blockDim.x + threadIdx.x;
  if (i >= NH) return;
  long n = i >> 7; int c = (int)(i & 127);
  long base = n * 384;
  float r  = gi[base + c]       + gh[base + c];
  float z  = gi[base + 128 + c] + gh[base + 128 + c];
  float nn = gi[base + 256 + c];
  float hn = gh[base + 256 + c];
  r = 1.f / (1.f + expf(-r));
  z = 1.f / (1.f + expf(-z));
  nn = tanhf(nn + r * hn);
  float h = hf[i];
  float o = (1.f - z) * nn + z * h;
  hf[i] = o; hb[i] = (bf16_t)o;
}

// ---------------- attention fusion over 4 views ----------------
__global__ __launch_bounds__(128) void attn_fuse_kernel(
    const float* __restrict__ qf, const float* __restrict__ keys,
    const float* __restrict__ hs, float* __restrict__ fused)
{
  const int n = blockIdx.x, t = threadIdx.x;
  __shared__ float part[4][128];
  __shared__ float aw[4];
  float qv = tanhf(qf[(long)n * HCH + t]);
  #pragma unroll
  for (int v = 0; v < 4; ++v)
    part[v][t] = qv * keys[((long)n * 4 + v) * HCH + t];
  __syncthreads();
  if (t < 4) {
    float s = 0.f;
    for (int k = 0; k < 128; ++k) s += part[t][k];
    aw[t] = s * 0.08838834764831845f;  // H^-0.5
  }
  __syncthreads();
  if (t == 0) {
    float mx = fmaxf(fmaxf(aw[0], aw[1]), fmaxf(aw[2], aw[3]));
    float e0 = expf(aw[0] - mx), e1 = expf(aw[1] - mx), e2 = expf(aw[2] - mx), e3 = expf(aw[3] - mx);
    float s = e0 + e1 + e2 + e3;
    aw[0] = e0 / s; aw[1] = e1 / s; aw[2] = e2 / s; aw[3] = e3 / s;
  }
  __syncthreads();
  float acc = 0.f;
  #pragma unroll
  for (int v = 0; v < 4; ++v) acc += aw[v] * hs[((long)n * 4 + v) * HCH + t];
  fused[(long)n * HCH + t] = acc;
}

// ---------------- pooling ----------------
__device__ inline void atomicMaxF(float* a, float v) {
  if (v >= 0.f) atomicMax((int*)a, __float_as_int(v));
  else          atomicMin((unsigned int*)a, __float_as_uint(v));
}

__global__ void pool_accum_kernel(const float* __restrict__ fused, const int* __restrict__ batch,
    float* __restrict__ psum, float* __restrict__ pmax, float* __restrict__ pcnt, long NH)
{
  long i = (long)blockIdx.x * blockDim.x + threadIdx.x;
  if (i >= NH) return;
  long n = i >> 7; int c = (int)(i & 127);
  int b = batch[n];
  float v = fused[i];
  atomicAdd(&psum[(long)b * HCH + c], v);
  atomicMaxF(&pmax[(long)b * HCH + c], v);
  if (c == 0) atomicAdd(&pcnt[b], 1.f);
}

__global__ void pool_final_kernel(const float* __restrict__ psum, const float* __restrict__ pmax,
                                  const float* __restrict__ pcnt, float* __restrict__ g, int BH)
{
  int i = blockIdx.x * blockDim.x + threadIdx.x;
  if (i >= BH) return;
  int b = i >> 7, c = i & 127;
  float cnt = fmaxf(pcnt[b], 1.f);
  g[(long)b * 256 + c]       = psum[i] / cnt;
  g[(long)b * 256 + 128 + c] = pmax[i];
}

// ---------------- classifier MLP: 256 -> 128 -> 64 -> 1, one block per graph ----------------
__global__ __launch_bounds__(128) void mlp_kernel(const float* __restrict__ g,
    const float* __restrict__ w1, const float* __restrict__ b1,
    const float* __restrict__ w2, const float* __restrict__ b2,
    const float* __restrict__ w3, const float* __restrict__ b3, float* __restrict__ out)
{
  const int b = blockIdx.x, t = threadIdx.x;
  __shared__ float sg[256], s1[128], s2[64];
  sg[t]       = g[(long)b * 256 + t];
  sg[t + 128] = g[(long)b * 256 + 128 + t];
  __syncthreads();
  float a = b1[t];
  for (int k = 0; k < 256; ++k) a += sg[k] * w1[k * 128 + t];
  s1[t] = fmaxf(a, 0.f);
  __syncthreads();
  if (t < 64) {
    float a2 = b2[t];
    for (int k = 0; k < 128; ++k) a2 += s1[k] * w2[k * 64 + t];
    s2[t] = fmaxf(a2, 0.f);
  }
  __syncthreads();
  if (t == 0) {
    float a3 = b3[0];
    for (int k = 0; k < 64; ++k) a3 += s2[k] * w3[k];
    out[b] = a3;
  }
}

// ---------------- host launch ----------------
extern "C" void kernel_launch(void* const* d_in, const int* in_sizes, int n_in,
                              void* d_out, int out_size, void* d_ws, size_t ws_size,
                              hipStream_t stream)
{
  const int N = 50000, E = 600000, IN = 512, H = 128, B = 64, L = 4;
  const float* x        = (const float*)d_in[0];
  const int*   eidx     = (const int*)d_in[1];
  const int*   etype    = (const int*)d_in[2];
  const int*   batch    = (const int*)d_in[3];
  const float* proj_w   = (const float*)d_in[4];
  const float* proj_b   = (const float*)d_in[5];
  const float* ln0_w    = (const float*)d_in[6];
  const float* ln0_b    = (const float*)d_in[7];
  const float* ggnn_w   = (const float*)d_in[8];
  const float* gru_wih  = (const float*)d_in[9];
  const float* gru_whh  = (const float*)d_in[10];
  const float* gru_bih  = (const float*)d_in[11];
  const float* gru_bhh  = (const float*)d_in[12];
  const float* vln_w    = (const float*)d_in[13];
  const float* vln_b    = (const float*)d_in[14];
  const float* q_w      = (const float*)d_in[15];
  const float* q_b      = (const float*)d_in[16];
  const float* k_w      = (const float*)d_in[17];
  const float* k_b      = (const float*)d_in[18];
  const float* c1_w     = (const float*)d_in[19];
  const float* c1_b     = (const float*)d_in[20];
  const float* c2_w     = (const float*)d_in[21];
  const float* c2_b     = (const float*)d_in[22];
  const float* c3_w     = (const float*)d_in[23];
  const float* c3_b     = (const float*)d_in[24];
  float* out = (float*)d_out;
  (void)in_sizes; (void)n_in; (void)out_size; (void)ws_size;

  const long NH  = (long)N * H;
  const long N3H = (long)N * 3 * H;
  const long N4H = (long)N * 4 * H;
  const long NIN = (long)N * IN;

  char* ws = (char*)d_ws;
  size_t off = 0;
  auto alloc = [&](size_t bytes) -> char* {
    char* p = ws + off;
    off += (bytes + 255) & ~(size_t)255;
    return p;
  };

  float*  hsf   = (float*)alloc(N4H * 4);          // [N,4,H] view outputs (f32)
  bf16_t* xb    = (bf16_t*)hsf;                    // alias: x bf16 dead before hsf written
  float*  gigh  = (float*)alloc(2 * N3H * 4);      // gi | gh
  float*  gif   = gigh;
  float*  ghf   = gigh + N3H;
  float*  keysf = gigh;                            // alias: keys (N4H f32 <= 2*N3H f32)
  float*  h0f   = (float*)alloc(NH * 4);
  float*  hf    = (float*)alloc(NH * 4);
  float*  aggf  = (float*)alloc(NH * 4);
  float*  scrA  = (float*)alloc(NH * 4);           // proj pre-LN, later q (pre-tanh)
  float*  fused = (float*)alloc(NH * 4);
  bf16_t* h0b   = (bf16_t*)alloc(NH * 2);
  bf16_t* hb    = (bf16_t*)alloc(NH * 2);
  bf16_t* aggb  = (bf16_t*)alloc(NH * 2);
  bf16_t* mb    = (bf16_t*)alloc(NH * 2);
  bf16_t* hsb   = (bf16_t*)alloc(N4H * 2);
  bf16_t* pwp   = (bf16_t*)alloc((size_t)IN * H * 2);        // packed weights
  bf16_t* ggp   = (bf16_t*)alloc((size_t)16 * H * H * 2);
  bf16_t* wihp  = (bf16_t*)alloc((size_t)4 * H * 3 * H * 2);
  bf16_t* whhp  = (bf16_t*)alloc((size_t)4 * H * 3 * H * 2);
  bf16_t* qwp   = (bf16_t*)alloc((size_t)H * H * 2);
  bf16_t* kwp   = (bf16_t*)alloc((size_t)H * H * 2);
  float*  psum  = (float*)alloc((size_t)B * H * 4);
  float*  pmax  = (float*)alloc((size_t)B * H * 4);
  float*  pcnt  = (float*)alloc((size_t)B * 4);
  float*  gpool = (float*)alloc((size_t)B * 2 * H * 4);

  auto cdiv = [](long a, long b) { return (int)((a + b - 1) / b); };

  // ---- input conversion + weight packing ----
  cvt_bf16_kernel<<<cdiv(NIN, 256), 256, 0, stream>>>(x, xb, NIN);
  pack_b_kernel<<<cdiv((long)IN * H, 256), 256, 0, stream>>>(proj_w, pwp, IN, H, 0);
  for (int j = 0; j < 16; ++j)
    pack_b_kernel<<<cdiv((long)H * H, 256), 256, 0, stream>>>(
        ggnn_w + (long)j * H * H, ggp + (long)j * H * H, H, H, 0);
  for (int v = 0; v < 4; ++v) {
    pack_b_kernel<<<cdiv((long)3 * H * H, 256), 256, 0, stream>>>(
        gru_wih + (long)v * 3 * H * H, wihp + (long)v * H * 3 * H, H, 3 * H, 1);
    pack_b_kernel<<<cdiv((long)3 * H * H, 256), 256, 0, stream>>>(
        gru_whh + (long)v * 3 * H * H, whhp + (long)v * H * 3 * H, H, 3 * H, 1);
  }
  pack_b_kernel<<<cdiv((long)H * H, 256), 256, 0, stream>>>(q_w, qwp, H, H, 0);
  pack_b_kernel<<<cdiv((long)H * H, 256), 256, 0, stream>>>(k_w, kwp, H, H, 0);

  // ---- input projection -> LN -> GELU ----
  gemm_wmma_bf16_kernel<<<dim3(cdiv(N, 64), 1), 256, 0, stream>>>(xb, pwp, proj_b, scrA, nullptr, N, IN, H);
  ln_kernel<<<N, 128, 0, stream>>>(scrA, ln0_w, ln0_b, nullptr, h0f, h0b, H, 1);

  // ---- 4 views x 4 GatedGraphConv layers ----
  const unsigned vmask[4] = {0x601u, 0xEu, 0x70u, 0x1980u};  // {0,9,10},{1,2,3},{4,5,6},{7,8,11,12}
  const int* src = eidx;
  const int* dst = eidx + E;
  for (int v = 0; v < 4; ++v) {
    hipMemcpyAsync(hf, h0f, NH * 4, hipMemcpyDeviceToDevice, stream);
    hipMemcpyAsync(hb, h0b, NH * 2, hipMemcpyDeviceToDevice, stream);
    for (int i = 0; i < L; ++i) {
      // m = h @ W[v,i]  (bf16 messages for the gather)
      gemm_wmma_bf16_kernel<<<dim3(cdiv(N, 64), 1), 256, 0, stream>>>(
          hb, ggp + (long)(v * L + i) * H * H, nullptr, nullptr, mb, N, H, H);
      hipMemsetAsync(aggf, 0, NH * 4, stream);
      scatter_kernel<<<cdiv((long)E * 32, 256), 256, 0, stream>>>(mb, src, dst, etype, vmask[v], aggf, E);
      cvt_bf16_kernel<<<cdiv(NH, 256), 256, 0, stream>>>(aggf, aggb, NH);
      // gi = agg @ wih^T + bih ; gh = h @ whh^T + bhh
      gemm_wmma_bf16_kernel<<<dim3(cdiv(N, 64), 3), 256, 0, stream>>>(
          aggb, wihp + (long)v * H * 3 * H, gru_bih + (long)v * 3 * H, gif, nullptr, N, H, 3 * H);
      gemm_wmma_bf16_kernel<<<dim3(cdiv(N, 64), 3), 256, 0, stream>>>(
          hb, whhp + (long)v * H * 3 * H, gru_bhh + (long)v * 3 * H, ghf, nullptr, N, H, 3 * H);
      gru_kernel<<<cdiv(NH, 256), 256, 0, stream>>>(gif, ghf, hf, hb, NH);
    }
    // views[v] = LN(h)*w+b + h0, written strided into hs [N,4,H]
    ln_kernel<<<N, 128, 0, stream>>>(hf, vln_w + v * H, vln_b + v * H, h0f,
                                     hsf + (long)v * H, hsb + (long)v * H, 4 * H, 0);
  }

  // ---- attention fusion ----
  gemm_wmma_bf16_kernel<<<dim3(cdiv(N, 64), 1), 256, 0, stream>>>(h0b, qwp, q_b, scrA, nullptr, N, H, H);
  gemm_wmma_bf16_kernel<<<dim3(cdiv(4 * N, 64), 1), 256, 0, stream>>>(hsb, kwp, k_b, keysf, nullptr, 4 * N, H, H);
  attn_fuse_kernel<<<N, 128, 0, stream>>>(scrA, keysf, hsf, fused);

  // ---- pooling + classifier ----
  fill_kernel<<<cdiv((long)B * H, 256), 256, 0, stream>>>(psum, 0.f, (long)B * H);
  fill_kernel<<<cdiv((long)B * H, 256), 256, 0, stream>>>(pmax, -3.402823466e38f, (long)B * H);
  fill_kernel<<<cdiv((long)B, 256), 256, 0, stream>>>(pcnt, 0.f, (long)B);
  pool_accum_kernel<<<cdiv(NH, 256), 256, 0, stream>>>(fused, batch, psum, pmax, pcnt, NH);
  pool_final_kernel<<<cdiv((long)B * H, 256), 256, 0, stream>>>(psum, pmax, pcnt, gpool, B * H);
  mlp_kernel<<<B, 128, 0, stream>>>(gpool, c1_w, c1_b, c2_w, c2_b, c3_w, c3_b, out);
}
